// NeuralDecisionTree_55336358642025
// MI455X (gfx1250) — matrix operand order repaired and checked by reading, not compiled, not run on hardware
//
#include <hip/hip_runtime.h>
#include <hip/hip_bf16.h>

// ---------------------------------------------------------------------------
// Neural decision tree, fully fused for MI455X (gfx1250, wave32, WMMA).
//
//  Prep (tiny):
//    - wp_frag: W rows scattered into feature-indexed bf16 weights, stored in
//      WMMA B-fragment layout (4 MB, L2-resident).
//    - probs_frag: softmax(pi) padded to 128 classes, bf16 B-fragment layout.
//  Main (512 threads = 16 waves, 32 batch rows per block):
//    Phase 1: d = sigmoid(features @ Wp + b)  [32 x 1024] tile kept in LDS,
//             64 K-steps, ping-pong A-fragments (one barrier per step),
//             wave (wm, wn) owns M-subtile wm and 8 N-subtiles.
//    Phase 2: tree routing (f32 VALU), mu -> LDS A-fragments.
//    Phase 3: out = mu @ softmax(pi), wave = (M-subtile, class-subtile).
// ---------------------------------------------------------------------------

typedef __attribute__((ext_vector_type(16))) __bf16 v16bf;
typedef __attribute__((ext_vector_type(8)))  float  v8f;

#define NDT_BATCH 16384
#define NDT_NF    2048   // total features (K of GEMM1)
#define NDT_NN    1024   // nodes / leaves (N of GEMM1, K of GEMM2)
#define NDT_NC    100    // classes
#define NDT_MT    32     // batch rows per block

union FragB { v16bf v; uint4 q[2]; };

static __device__ __forceinline__ unsigned int pack_bf16x2(float x, float y) {
  unsigned int ux = __builtin_bit_cast(unsigned int, x);
  unsigned int uy = __builtin_bit_cast(unsigned int, y);
  ux = (ux + 0x7FFFu + ((ux >> 16) & 1u)) >> 16;
  uy = (uy + 0x7FFFu + ((uy >> 16) & 1u)) >> 16;
  return (uy << 16) | (ux & 0xFFFFu);
}
static __device__ __forceinline__ unsigned short f2bf(float x) {
  unsigned int ux = __builtin_bit_cast(unsigned int, x);
  return (unsigned short)((ux + 0x7FFFu + ((ux >> 16) & 1u)) >> 16);
}

// -------------------- prep: zero workspace region ---------------------------
__global__ void ndt_zero_kernel(unsigned int* __restrict__ p, int n_dwords) {
  int i = blockIdx.x * blockDim.x + threadIdx.x;
  int stride = gridDim.x * blockDim.x;
  for (; i < n_dwords; i += stride) p[i] = 0u;
}

// -------------------- prep: scatter W into Wp B-fragments -------------------
// B-fragment layout used here (contiguous k, 16-bit, 32x16 K x N): lane = (k>=16)*16 + n%16,
// VGPR v = (k%16)>>1, half = k&1. Block layout: [(T*64+s)*32 + lane] * 16 u16.
__global__ void ndt_scatter_w_kernel(const float* __restrict__ mask,
                                     const float* __restrict__ W,
                                     unsigned short* __restrict__ wp_frag) {
  __shared__ int s_idx;
  const int j = blockIdx.x;    // used-feature row, 0..1023
  const int t = threadIdx.x;   // 256 threads
  if (t == 0) s_idx = 0;
  __syncthreads();
  for (int c = t; c < NDT_NF; c += 256)
    if (mask[(size_t)j * NDT_NF + c] > 0.5f) s_idx = c;   // one-hot: one writer
  __syncthreads();
  const int f  = s_idx;        // feature index (K of GEMM1)
  const int T  = f >> 5;       // K-step
  const int kp = f & 31;
  const int v    = (kp & 15) >> 1;
  const int half = kp & 1;
  const int lhi  = (kp >> 4) << 4;
  for (int n = t; n < NDT_NN; n += 256) {
    int s = n >> 4;
    int lane = lhi + (n & 15);
    size_t idx = ((((size_t)(T * 64 + s)) * 32 + lane) * 8 + v) * 2 + half;
    wp_frag[idx] = f2bf(W[(size_t)j * NDT_NN + n]);
  }
}

// -------------------- prep: softmax(pi) into probs B-fragments --------------
__global__ void ndt_softmax_kernel(const float* __restrict__ pi,
                                   unsigned short* __restrict__ probs_frag) {
  __shared__ float red[128];
  const int k = blockIdx.x;    // leaf (K of GEMM2)
  const int t = threadIdx.x;   // 128 threads (class, padded)
  float x = (t < NDT_NC) ? pi[(size_t)k * NDT_NC + t] : -1e30f;
  red[t] = x;
  __syncthreads();
  for (int o = 64; o > 0; o >>= 1) { if (t < o) red[t] = fmaxf(red[t], red[t + o]); __syncthreads(); }
  float mx = red[0];
  __syncthreads();
  float e = (t < NDT_NC) ? __expf(x - mx) : 0.0f;
  red[t] = e;
  __syncthreads();
  for (int o = 64; o > 0; o >>= 1) { if (t < o) red[t] += red[t + o]; __syncthreads(); }
  float p = e / red[0];
  const int T  = k >> 5;
  const int kp = k & 31;
  const int v    = (kp & 15) >> 1;
  const int half = kp & 1;
  const int lane = ((kp >> 4) << 4) + (t & 15);
  const int s    = t >> 4;
  probs_frag[((((size_t)(T * 8 + s)) * 32 + lane) * 8 + v) * 2 + half] = f2bf(p);
}

// -------------------- main fused kernel -------------------------------------
__global__ __launch_bounds__(512)
void ndt_main_kernel(const float* __restrict__ features,
                     const float* __restrict__ bias,
                     const unsigned short* __restrict__ wp_frag,
                     const unsigned short* __restrict__ probs_frag,
                     float* __restrict__ out) {
  __shared__ float        d_lds[NDT_MT][NDT_NN + 8];   // 132 KB
  __shared__ unsigned int a_frag[2][512];              // 4 KB, ping-pong, 2 M-subtiles
  __shared__ unsigned int mu_frag[32 * 512];           // 64 KB (mu A-fragments)

  const int tid  = threadIdx.x;
  const int lane = tid & 31;
  const int wave = tid >> 5;     // 0..15
  const int wm   = wave >> 3;    // M-subtile 0..1
  const int wn   = wave & 7;     // N-subtile group 0..7
  const int row0 = blockIdx.x * NDT_MT;
  const int nlo  = lane & 15;
  const int chi  = lane >> 4;

  // ---- Phase 1: d = sigmoid(features @ Wp + b), tile [32 x 1024] ----------
  v8f acc[8];
  #pragma unroll
  for (int j = 0; j < 8; ++j) {
    v8f z = {0.f, 0.f, 0.f, 0.f, 0.f, 0.f, 0.f, 0.f};
    acc[j] = z;
  }

  // A-fragment write slot: 512 threads, one dword each, covering 32 rows x 16
  // k-pairs. A layout used here (contiguous k, 16x32, MxK): lane = (k/16)*16 + m%16,
  // v = (k%16)>>1.
  const int am   = tid >> 4;          // row 0..31
  const int ak   = (tid & 15) << 1;   // even k 0..30
  const int akk  = ak & 15;
  const int a_v    = akk >> 1;
  const int a_lane = ((ak >> 4) << 4) + (am & 15);
  const int a_dst  = (am >> 4) * 256 + a_lane * 8 + a_v;
  const float* a_src = features + (size_t)(row0 + am) * NDT_NF + ak;

  // Prologue: stage T=0 into buffer 0.
  {
    float2 f2 = *(const float2*)(a_src);
    a_frag[0][a_dst] = pack_bf16x2(f2.x, f2.y);
  }
  __syncthreads();

  for (int T = 0; T < 64; ++T) {
    const int cur = T & 1;
    // Stage next tile into the other buffer (overlaps with WMMAs below).
    if (T + 1 < 64) {
      float2 f2 = *(const float2*)(a_src + (T + 1) * 32);
      a_frag[cur ^ 1][a_dst] = pack_bf16x2(f2.x, f2.y);
    }
    FragB a;
    a.q[0] = *(const uint4*)&a_frag[cur][wm * 256 + lane * 8];
    a.q[1] = *(const uint4*)&a_frag[cur][wm * 256 + lane * 8 + 4];
    #pragma unroll
    for (int j = 0; j < 8; ++j) {
      int s = wn * 8 + j;
      const uint4* bp =
          (const uint4*)(wp_frag + ((size_t)(T * 64 + s) * 32 + lane) * 16);
      FragB b;
      b.q[0] = bp[0];
      b.q[1] = bp[1];
      acc[j] = __builtin_amdgcn_wmma_f32_16x16x32_bf16(
          false, a.v, false, b.v, (short)0, acc[j], false, false);
    }
    // One barrier per step: makes next-buffer stores visible and guarantees
    // this buffer's reads are done before it is overwritten at T+2.
    __syncthreads();
  }

  // Epilogue: bias + sigmoid into LDS. C layout: element (m = v+8*hi, n=lane%16).
  #pragma unroll
  for (int j = 0; j < 8; ++j) {
    int n = (wn * 8 + j) * 16 + nlo;
    float bv = bias[n];
    #pragma unroll
    for (int v = 0; v < 8; ++v) {
      float t = acc[j][v] + bv;
      d_lds[wm * 16 + v + 8 * chi][n] = 1.0f / (1.0f + __expf(-t));
    }
  }
  __syncthreads();

  // ---- Phase 2: tree routing, mu -> A-fragments in LDS ---------------------
  // (row, leaf-pair): leaves 2*pr, 2*pr+1 share the first 9 ancestors.
  for (int i = tid; i < NDT_MT * 512; i += 512) {
    int m  = i >> 9;           // row 0..31
    int pr = i & 511;          // leaf pair
    float p = 1.0f;
    #pragma unroll
    for (int k = 0; k < 9; ++k) {
      int node = (1 << k) + (pr >> (9 - k));
      int bit  = (pr >> (8 - k)) & 1;
      float d  = d_lds[m][node];
      p *= bit ? (1.0f - d) : d;
    }
    float d9  = d_lds[m][512 + pr];
    float mu0 = p * d9;
    float mu1 = p * (1.0f - d9);
    int leaf = pr << 1;
    int T  = leaf >> 5;
    int kp = leaf & 31;
    int kk = kp & 15;
    int v  = kk >> 1;
    int ln = ((kp >> 4) << 4) + (m & 15);
    mu_frag[(T * 2 + (m >> 4)) * 256 + ln * 8 + v] = pack_bf16x2(mu0, mu1);
  }
  __syncthreads();

  // ---- Phase 3: out = mu @ softmax(pi) -------------------------------------
  // wave = (wm = M-subtile, wn = class-subtile): exactly 16 tiles.
  v8f acc2 = {0.f, 0.f, 0.f, 0.f, 0.f, 0.f, 0.f, 0.f};
  for (int T = 0; T < 32; ++T) {
    FragB a;
    a.q[0] = *(const uint4*)&mu_frag[(T * 2 + wm) * 256 + lane * 8];
    a.q[1] = *(const uint4*)&mu_frag[(T * 2 + wm) * 256 + lane * 8 + 4];
    const uint4* bp =
        (const uint4*)(probs_frag + ((size_t)(T * 8 + wn) * 32 + lane) * 16);
    FragB b;
    b.q[0] = bp[0];
    b.q[1] = bp[1];
    acc2 = __builtin_amdgcn_wmma_f32_16x16x32_bf16(
        false, a.v, false, b.v, (short)0, acc2, false, false);
  }
  int col = wn * 16 + nlo;
  if (col < NDT_NC) {
    #pragma unroll
    for (int v = 0; v < 8; ++v) {
      int row = row0 + wm * 16 + v + 8 * chi;
      out[(size_t)row * NDT_NC + col] = acc2[v];
    }
  }
}

// -------------------- launch -------------------------------------------------
extern "C" void kernel_launch(void* const* d_in, const int* in_sizes, int n_in,
                              void* d_out, int out_size, void* d_ws, size_t ws_size,
                              hipStream_t stream) {
  const float* features = (const float*)d_in[0];  // [16384, 2048]
  const float* mask     = (const float*)d_in[1];  // [1024, 2048]
  const float* W        = (const float*)d_in[2];  // [1024, 1024]
  const float* bias     = (const float*)d_in[3];  // [1024]
  const float* pi       = (const float*)d_in[4];  // [1024, 100]
  float* out = (float*)d_out;                     // [16384, 100]

  unsigned short* wp_frag    = (unsigned short*)d_ws;             // 4 MB
  unsigned short* probs_frag = wp_frag + (size_t)NDT_NF * NDT_NN; // 256 KB

  ndt_zero_kernel<<<512, 256, 0, stream>>>((unsigned int*)wp_frag,
                                           (NDT_NF * NDT_NN) / 2);
  ndt_scatter_w_kernel<<<NDT_NN, 256, 0, stream>>>(mask, W, wp_frag);
  ndt_softmax_kernel<<<NDT_NN, 128, 0, stream>>>(pi, probs_frag);
  ndt_main_kernel<<<NDT_BATCH / NDT_MT, 512, 0, stream>>>(features, bias, wp_frag,
                                                          probs_frag, out);
}
